// VectorQuantizer_28518582845581
// MI455X (gfx1250) — matrix operand order repaired and checked by reading, not compiled, not run on hardware
//
#include <hip/hip_runtime.h>
#include <hip/hip_bf16.h>

typedef __attribute__((ext_vector_type(16))) __bf16 v16bf;
typedef __attribute__((ext_vector_type(8)))  float  v8f;

#define N_CODES   8192
#define CODE_DIM  512
#define N_ROWS    16384           // 16 * 1024
#define ZQ_ELEMS  (N_ROWS * CODE_DIM)   // 8388608
#define A_STRIDE  520             // 512 + 8 ushort pad -> 1040B row stride (bank-conflict free)

// ---------- helpers ----------
__device__ __forceinline__ unsigned short f2bf(float f) {
    unsigned int u = __float_as_uint(f);
    unsigned int r = u + 0x7fffu + ((u >> 16) & 1u);   // round-to-nearest-even
    return (unsigned short)(r >> 16);
}
__device__ __forceinline__ float bf2f(unsigned short h) {
    return __uint_as_float(((unsigned int)h) << 16);
}

__device__ __forceinline__ float block_reduce_sum(float v) {
    #pragma unroll
    for (int off = 16; off > 0; off >>= 1) v += __shfl_down(v, off, 32);
    __shared__ float sm[8];
    int lane = threadIdx.x & 31, wave = threadIdx.x >> 5;
    if (lane == 0) sm[wave] = v;
    __syncthreads();
    float r = 0.0f;
    if (threadIdx.x < 8) r = sm[threadIdx.x];
    if (threadIdx.x < 32) {
        #pragma unroll
        for (int off = 4; off > 0; off >>= 1) r += __shfl_down(r, off, 32);
    }
    return r;   // valid in thread 0
}

// ---------- kernel 1: codebook -> bf16 hi/lo split + row norms ----------
__global__ __launch_bounds__(256) void vq_conv_codebook(
    const float* __restrict__ cb,
    unsigned short* __restrict__ cb_hi, unsigned short* __restrict__ cb_lo,
    float* __restrict__ cnorm)
{
    int j = blockIdx.x;               // code row
    int t = threadIdx.x;
    float sq = 0.0f;
    #pragma unroll
    for (int e = 0; e < 2; ++e) {
        int k = t + e * 256;
        float x = cb[(size_t)j * CODE_DIM + k];
        unsigned short h = f2bf(x);
        unsigned short l = f2bf(x - bf2f(h));
        cb_hi[(size_t)j * CODE_DIM + k] = h;
        cb_lo[(size_t)j * CODE_DIM + k] = l;
        sq += x * x;
    }
    float s = block_reduce_sum(sq);
    if (t == 0) cnorm[j] = s;
}

// ---------- kernel 2: z_e -> bf16 hi/lo split (staged in d_out scratch) ----------
__global__ __launch_bounds__(256) void vq_conv_z(
    const float4* __restrict__ z4,
    uint2* __restrict__ ze_hi4, uint2* __restrict__ ze_lo4)
{
    int i = blockIdx.x * 256 + threadIdx.x;   // covers ZQ_ELEMS/4 vec4s
    float4 x = z4[i];
    unsigned short h0 = f2bf(x.x), h1 = f2bf(x.y), h2 = f2bf(x.z), h3 = f2bf(x.w);
    unsigned short l0 = f2bf(x.x - bf2f(h0));
    unsigned short l1 = f2bf(x.y - bf2f(h1));
    unsigned short l2 = f2bf(x.z - bf2f(h2));
    unsigned short l3 = f2bf(x.w - bf2f(h3));
    uint2 hp, lp;
    hp.x = (unsigned)h0 | ((unsigned)h1 << 16);
    hp.y = (unsigned)h2 | ((unsigned)h3 << 16);
    lp.x = (unsigned)l0 | ((unsigned)l1 << 16);
    lp.y = (unsigned)l2 | ((unsigned)l3 << 16);
    ze_hi4[i] = hp;
    ze_lo4[i] = lp;
}

// ---------- kernel 3: fused GEMM (x . c^T) + running argmin ----------
union Frag { uint4 q[2]; v16bf v; };

__global__ __launch_bounds__(256) void vq_argmin(
    const unsigned short* __restrict__ ze_hi, const unsigned short* __restrict__ ze_lo,
    const unsigned short* __restrict__ cb_hi, const unsigned short* __restrict__ cb_lo,
    const float* __restrict__ cnorm, int* __restrict__ idx_out)
{
    __shared__ unsigned short a_hi[32 * A_STRIDE];   // 33.3 KB, padded rows
    __shared__ unsigned short a_lo[32 * A_STRIDE];
    __shared__ float s_val[8][32];
    __shared__ int   s_idx[8][32];

    const int lane  = threadIdx.x & 31;
    const int wave  = threadIdx.x >> 5;
    const int m_blk = blockIdx.x * 32;          // this block's 32 rows
    const int col   = lane & 15;
    const int kA    = (lane & 16) ? 8  : 0;     // A-frag lane-half K offset
    const int kB    = (lane & 16) ? 8  : 0;     // B-frag lane-half K offset

    // ---- cooperative stage of the block's 32x512 hi/lo A tile into LDS ----
    {
        const uint4* gh = (const uint4*)(ze_hi + (size_t)m_blk * CODE_DIM);
        const uint4* gl = (const uint4*)(ze_lo + (size_t)m_blk * CODE_DIM);
        #pragma unroll
        for (int c = threadIdx.x; c < 32 * (CODE_DIM / 8); c += 256) {
            int row = c >> 6;              // 64 uint4 chunks per row
            int within = c & 63;
            uint4* lh = (uint4*)(a_hi + row * A_STRIDE + within * 8);
            uint4* ll = (uint4*)(a_lo + row * A_STRIDE + within * 8);
            *lh = gh[c];
            *ll = gl[c];
        }
    }
    __syncthreads();

    float bestv[2][8];
    int   bestn[2][8];
    #pragma unroll
    for (int mi = 0; mi < 2; ++mi)
        #pragma unroll
        for (int r = 0; r < 8; ++r) { bestv[mi][r] = 3.4e38f; bestn[mi][r] = 0; }

    #pragma unroll 1
    for (int n_iter = 0; n_iter < N_CODES / 256; ++n_iter) {
        const int n0 = n_iter * 256 + wave * 32;    // this wave's 32 codes

        // pull next code tile toward the WGP while this one computes (L2-resident)
        {
            int nxt = n0 + 256;
            if (nxt >= N_CODES) nxt = n0;
            __builtin_prefetch(cb_hi + (size_t)(nxt + lane) * CODE_DIM, 0, 1);
            __builtin_prefetch(cb_lo + (size_t)(nxt + lane) * CODE_DIM, 0, 1);
        }

        v8f acc[2][2];
        #pragma unroll
        for (int mi = 0; mi < 2; ++mi)
            #pragma unroll
            for (int ni = 0; ni < 2; ++ni) acc[mi][ni] = (v8f){};

        #pragma unroll 1
        for (int kk = 0; kk < CODE_DIM; kk += 32) {
            Frag ah[2], al[2], bh[2], bl[2];
            #pragma unroll
            for (int ni = 0; ni < 2; ++ni) {
                size_t off = (size_t)(n0 + ni * 16 + col) * CODE_DIM + kk + kB;
                const unsigned short* ph = cb_hi + off;
                const unsigned short* pl = cb_lo + off;
                bh[ni].q[0] = *(const uint4*)(ph);
                bh[ni].q[1] = *(const uint4*)(ph + 16);
                bl[ni].q[0] = *(const uint4*)(pl);
                bl[ni].q[1] = *(const uint4*)(pl + 16);
            }
            #pragma unroll
            for (int mi = 0; mi < 2; ++mi) {
                int off = (mi * 16 + col) * A_STRIDE + kk + kA;
                const unsigned short* ph = a_hi + off;
                const unsigned short* pl = a_lo + off;
                ah[mi].q[0] = *(const uint4*)(ph);
                ah[mi].q[1] = *(const uint4*)(ph + 16);
                al[mi].q[0] = *(const uint4*)(pl);
                al[mi].q[1] = *(const uint4*)(pl + 16);
            }
            #pragma unroll
            for (int mi = 0; mi < 2; ++mi)
                #pragma unroll
                for (int ni = 0; ni < 2; ++ni) {
                    // x.c ~= xh*ch + xh*cl + xl*ch  (fp32 accumulate)
                    acc[mi][ni] = __builtin_amdgcn_wmma_f32_16x16x32_bf16(
                        false, ah[mi].v, false, bh[ni].v, (short)0, acc[mi][ni], false, false);
                    acc[mi][ni] = __builtin_amdgcn_wmma_f32_16x16x32_bf16(
                        false, ah[mi].v, false, bl[ni].v, (short)0, acc[mi][ni], false, false);
                    acc[mi][ni] = __builtin_amdgcn_wmma_f32_16x16x32_bf16(
                        false, al[mi].v, false, bh[ni].v, (short)0, acc[mi][ni], false, false);
                }
        }

        // score = cnorm[n] - 2*dot ; branchless running argmin.
        // Per lane the visited code index is strictly increasing, so strict '<'
        // keeps the earliest (lowest-index) minimum — matching jnp.argmin.
        #pragma unroll
        for (int ni = 0; ni < 2; ++ni) {
            int n_col = n0 + ni * 16 + col;
            float cn = cnorm[n_col];
            #pragma unroll
            for (int mi = 0; mi < 2; ++mi)
                #pragma unroll
                for (int r = 0; r < 8; ++r) {
                    float s = __builtin_fmaf(-2.0f, acc[mi][ni][r], cn);
                    bestn[mi][r] = (s < bestv[mi][r]) ? n_col : bestn[mi][r];
                    bestv[mi][r] = fminf(s, bestv[mi][r]);
                }
        }
    }

    // reduce across the 16 lanes of each half (columns); lowest index wins ties
    #pragma unroll
    for (int off = 1; off < 16; off <<= 1) {
        #pragma unroll
        for (int mi = 0; mi < 2; ++mi)
            #pragma unroll
            for (int r = 0; r < 8; ++r) {
                float ov = __shfl_xor(bestv[mi][r], off, 32);
                int   on = __shfl_xor(bestn[mi][r], off, 32);
                int take = (int)(ov < bestv[mi][r]) |
                           ((int)(ov == bestv[mi][r]) & (int)(on < bestn[mi][r]));
                bestv[mi][r] = take ? ov : bestv[mi][r];
                bestn[mi][r] = take ? on : bestn[mi][r];
            }
    }

    // cross-wave reduction via LDS (each wave saw a disjoint code subset)
    if ((lane & 15) == 0) {
        int half = lane >> 4;       // lanes<16 -> rows r; lanes>=16 -> rows r+8
        #pragma unroll
        for (int mi = 0; mi < 2; ++mi)
            #pragma unroll
            for (int r = 0; r < 8; ++r) {
                int row = mi * 16 + half * 8 + r;
                s_val[wave][row] = bestv[mi][r];
                s_idx[wave][row] = bestn[mi][r];
            }
    }
    __syncthreads();
    if (threadIdx.x < 32) {
        int row = threadIdx.x;
        float bv = s_val[0][row]; int bi = s_idx[0][row];
        #pragma unroll
        for (int w = 1; w < 8; ++w) {
            float ov = s_val[w][row]; int oi = s_idx[w][row];
            int take = (int)(ov < bv) | ((int)(ov == bv) & (int)(oi < bi));
            bv = take ? ov : bv;
            bi = take ? oi : bi;
        }
        idx_out[m_blk + row] = bi;
    }
}

// ---------- kernel 4: gather z_q, commit-loss partial sums, histogram ----------
__global__ __launch_bounds__(256) void vq_gather(
    const float* __restrict__ ze, const float* __restrict__ cb,
    const int* __restrict__ idx, float* __restrict__ zq_out,
    float* __restrict__ commit_sum, float* __restrict__ counts,
    float* __restrict__ idx_f)
{
    int row = blockIdx.x;
    int t = threadIdx.x;
    int j = idx[row];
    const float* c = cb + (size_t)j * CODE_DIM;
    const float* x = ze + (size_t)row * CODE_DIM;
    float local = 0.0f;
    #pragma unroll
    for (int e = 0; e < 2; ++e) {
        int k = t + e * 256;
        float cv = c[k], xv = x[k];
        zq_out[(size_t)row * CODE_DIM + k] = cv;   // straight-through fwd value == z_q
        float d = xv - cv;
        local += d * d;
    }
    float s = block_reduce_sum(local);
    if (t == 0) {
        atomicAdd(commit_sum, s);
        atomicAdd(&counts[j], 1.0f);
        idx_f[row] = (float)j;
    }
}

// ---------- kernel 5: scalars (commit, codebook_loss, perplexity) ----------
__global__ __launch_bounds__(256) void vq_finalize(
    const float* __restrict__ counts, const float* __restrict__ commit_sum,
    float* __restrict__ out_scalars)
{
    int t = threadIdx.x;
    const float total = (float)N_ROWS;
    float h = 0.0f;
    for (int j = t; j < N_CODES; j += 256) {
        float c = counts[j];
        if (c > 0.0f) {
            float p = c / total;
            h -= p * __logf(p);
        }
    }
    float H = block_reduce_sum(h);
    if (t == 0) {
        out_scalars[0] = 0.25f * commit_sum[0] / (float)ZQ_ELEMS;  // commit
        out_scalars[1] = 0.0f;                                     // codebook loss (ema path)
        out_scalars[2] = __expf(H);                                // perplexity
    }
}

extern "C" void kernel_launch(void* const* d_in, const int* in_sizes, int n_in,
                              void* d_out, int out_size, void* d_ws, size_t ws_size,
                              hipStream_t stream) {
    const float* z  = (const float*)d_in[0];   // [16,1024,512] f32
    const float* cb = (const float*)d_in[1];   // [8192,512] f32
    float* out = (float*)d_out;

    // workspace layout (~17 MB)
    char* ws = (char*)d_ws;
    unsigned short* cb_hi = (unsigned short*)(ws);
    unsigned short* cb_lo = (unsigned short*)(ws + (size_t)N_CODES * CODE_DIM * 2);
    char* p = ws + (size_t)N_CODES * CODE_DIM * 4;
    float* cnorm  = (float*)p;                 p += N_CODES * 4;
    int*   idx    = (int*)p;                   p += N_ROWS * 4;
    float* counts = (float*)p;                 p += N_CODES * 4;
    float* sums   = (float*)p;                 // commit-sum accumulator

    // bf16-split z_e staged inside d_out's z_q region (32 MB); overwritten by gather later
    unsigned short* ze_hi = (unsigned short*)d_out;
    unsigned short* ze_lo = (unsigned short*)((char*)d_out + (size_t)N_ROWS * CODE_DIM * 2);

    // zero histogram + accumulator (counts and sums are contiguous)
    hipMemsetAsync(counts, 0, (size_t)N_CODES * 4 + 256, stream);

    vq_conv_codebook<<<N_CODES, 256, 0, stream>>>(cb, cb_hi, cb_lo, cnorm);
    vq_conv_z<<<ZQ_ELEMS / 4 / 256, 256, 0, stream>>>(
        (const float4*)z, (uint2*)ze_hi, (uint2*)ze_lo);
    vq_argmin<<<N_ROWS / 32, 256, 0, stream>>>(ze_hi, ze_lo, cb_hi, cb_lo, cnorm, idx);
    vq_gather<<<N_ROWS, 256, 0, stream>>>(z, cb, idx, out, sums, counts,
                                          out + (size_t)ZQ_ELEMS + 3);
    vq_finalize<<<1, 256, 0, stream>>>(counts, sums, out + ZQ_ELEMS);
}
